// MsaHmmCell_58067957841936
// MI455X (gfx1250) — matrix operand order repaired and checked by reading, not compiled, not run on hardware
//
#include <hip/hip_runtime.h>

typedef __attribute__((ext_vector_type(16))) __bf16 v16bf;
typedef __attribute__((ext_vector_type(8)))  float  v8f;

#define LDIM 512
#define SY 26
#define NST 1027        // N = 2L+3
#define LNP 1056        // padded N (33*32 == 66*16)
#define NKT 33          // K tiles of 32
#define NNT 80          // padded N tiles of 16 (5 per wave * 16 waves)
#define NNT_REAL 66
#define UAI 1024
#define RFI 1025
#define TMI 1026

struct Probs {
  const float *b2m, *m2e, *m2m, *m2i, *i2m, *i2i, *md, *d2m, *cp, *entry, *exitp, *scal;
};
// scal: [0]=p_lfl [1]=p_lfe [2]=p_e2u [3]=p_e2r [4]=p_e2t [5]=B2E

// ---------------------------------------------------------------------------
// Dense value of implicit transition matrix A[i][j] (i,j in padded space)
// ---------------------------------------------------------------------------
__device__ float Aval(const Probs& P, int i, int j) {
  if (i >= NST || j >= NST) return 0.f;
  const float lfl = P.scal[0], lfe = P.scal[1];
  const float e2u = P.scal[2], e2r = P.scal[3], e2t = P.scal[4], B2E = P.scal[5];
  if (i == 0) {                                   // left flank
    if (j == 0)   return lfl;
    if (j >= 1 && j <= LDIM) return lfe * P.entry[j - 1];
    if (j == UAI) return lfe * B2E * e2u;
    if (j == RFI) return lfe * B2E * e2r;
    if (j == TMI) return lfe * B2E * e2t;
    return 0.f;
  }
  if (i <= LDIM) {                                // match state i
    if (i < LDIM && j == i + 1) return P.m2m[i - 1];
    if (j >= i + 2 && j <= LDIM) {                // match_skip via deletes
      int jp = j - 2;
      return P.md[i] * (P.cp[jp] / fmaxf(P.cp[i], 1e-30f)) * P.d2m[jp];
    }
    if (j == UAI) return P.exitp[i - 1] * e2u;
    if (j == RFI) return P.exitp[i - 1] * e2r;
    if (j == TMI) return P.exitp[i - 1] * e2t;
    if (i < LDIM && j == LDIM + i) return P.m2i[i - 1];  // insert i-1 at col L+i
    return 0.f;
  }
  if (i <= 2 * LDIM - 1) {                        // insert q = i-(L+1)
    int q = i - (LDIM + 1);
    if (j == q + 2) return P.i2m[q];
    if (j == i)     return P.i2i[q];
    return 0.f;
  }
  if (i == UAI) {                                 // unannotated segment
    if (j >= 1 && j <= LDIM) return lfe * P.entry[j - 1];
    if (j == UAI) return lfl + lfe * B2E * e2u;
    if (j == RFI) return lfe * B2E * e2r;
    if (j == TMI) return lfe * B2E * e2t;
    return 0.f;
  }
  if (i == RFI) return (j == RFI) ? lfl : ((j == TMI) ? lfe : 0.f);
  return (j == TMI) ? 1.0f : 0.f;                 // terminal
}

// ---------------------------------------------------------------------------
// S1: per-row softmax of explicit transition logits
// ---------------------------------------------------------------------------
__global__ void s1_probs(const float* b2m, const float* m2e, const float* m2m,
                         const float* m2i, const float* i2m, const float* i2i,
                         const float* md, const float* d2m, const float* d2d,
                         const float* lfl, const float* lfe,
                         const float* e2u, const float* e2r, const float* e2t,
                         float* P_b2m, float* P_m2e, float* P_m2m, float* P_m2i,
                         float* P_i2m, float* P_i2i, float* P_md, float* P_d2m,
                         float* P_d2d, float* scal) {
  int tid = threadIdx.x;
  if (tid == 0) {
    // flank rows (lf, ua, rf share logits)
    float a = lfl[0], b = lfe[0], mx = fmaxf(a, b);
    float ea = expf(a - mx), eb = expf(b - mx), z = ea + eb;
    scal[0] = ea / z; scal[1] = eb / z;
    // end row trio
    float u = e2u[0], r = e2r[0], t = e2t[0];
    mx = fmaxf(fmaxf(u, r), t);
    float eu = expf(u - mx), er = expf(r - mx), et = expf(t - mx);
    z = eu + er + et;
    scal[2] = eu / z; scal[3] = er / z; scal[4] = et / z;
    // begin row: L begin_to_match logits + match_to_delete[0]
    mx = md[0];
    for (int k = 0; k < LDIM; ++k) mx = fmaxf(mx, b2m[k]);
    z = expf(md[0] - mx);
    for (int k = 0; k < LDIM; ++k) z += expf(b2m[k] - mx);
    P_md[0] = expf(md[0] - mx) / z;
    for (int k = 0; k < LDIM; ++k) P_b2m[k] = expf(b2m[k] - mx) / z;
    P_m2e[LDIM - 1] = 1.0f;   // match row L: only edge
    P_d2m[LDIM - 1] = 1.0f;   // delete row L-1: only edge
  }
  if (tid >= 1 && tid <= LDIM - 1) {              // match row i = tid (4 edges)
    int i = tid;
    float l0 = m2e[i - 1], l1 = m2m[i - 1], l2 = m2i[i - 1], l3 = md[i];
    float mx = fmaxf(fmaxf(l0, l1), fmaxf(l2, l3));
    float e0 = expf(l0 - mx), e1 = expf(l1 - mx), e2 = expf(l2 - mx), e3 = expf(l3 - mx);
    float z = e0 + e1 + e2 + e3;
    P_m2e[i - 1] = e0 / z; P_m2m[i - 1] = e1 / z; P_m2i[i - 1] = e2 / z; P_md[i] = e3 / z;
  }
  if (tid < LDIM - 1) {                            // insert + delete pair rows
    float a = i2m[tid], b = i2i[tid], mx = fmaxf(a, b);
    float ea = expf(a - mx), eb = expf(b - mx), z = ea + eb;
    P_i2m[tid] = ea / z; P_i2i[tid] = eb / z;
    a = d2m[tid]; b = d2d[tid]; mx = fmaxf(a, b);
    ea = expf(a - mx); eb = expf(b - mx); z = ea + eb;
    P_d2m[tid] = ea / z; P_d2d[tid] = eb / z;
  }
}

// ---------------------------------------------------------------------------
// S2: cumprod of delete chain + entry/exit adds + B2E (tiny; single thread)
// ---------------------------------------------------------------------------
__global__ void s2_scan(const float* P_md, const float* P_d2m, const float* P_d2d,
                        const float* P_b2m, const float* P_m2e,
                        float* cp, float* entry, float* exitp, float* scal) {
  cp[0] = 1.0f;
  for (int k = 1; k < LDIM; ++k) cp[k] = cp[k - 1] * P_d2d[k - 1];
  float B2E = P_md[0] * cp[LDIM - 1] * P_d2m[LDIM - 1];
  scal[5] = B2E;
  entry[0] = P_b2m[0];
  for (int k = 1; k < LDIM; ++k)
    entry[k] = P_b2m[k] + P_md[0] * cp[k - 1] * P_d2m[k - 1];
  for (int i = 0; i < LDIM - 1; ++i)
    exitp[i] = P_m2e[i] + P_md[i + 1] * (cp[LDIM - 1] / fmaxf(cp[i + 1], 1e-30f)) * P_d2m[LDIM - 1];
  exitp[LDIM - 1] = P_m2e[LDIM - 1];
}

// ---------------------------------------------------------------------------
// S3: build A in bf16, pre-swizzled into WMMA B-operand (32x16) tile layout.
// Tile (kt,nt): element index = lane*16+e ; lanes 0-15 -> N=lane, K=e<8 ? e : 8+e ;
// lanes 16-31 -> N=lane-16, K=e<8 ? 8+e : 16+e   (ISA 7.12.2 16-bit B layout)
// ---------------------------------------------------------------------------
__global__ void s3_atile(Probs P, __bf16* Atile) {
  int nt = blockIdx.x, kt = blockIdx.y, tid = threadIdx.x;
  int lane = tid >> 4, e = tid & 15;
  int K = kt * 32 + ((lane >= 16) ? 8 : 0) + (e & 7) + ((e >= 8) ? 16 : 0);
  int j = nt * 16 + (lane & 15);
  Atile[(size_t)(kt * NNT + nt) * 512 + tid] = (__bf16)Aval(P, K, j);
}

// ---------------------------------------------------------------------------
// S4: BmT[s][j] = Bm[j][s] padded to LNP columns
// ---------------------------------------------------------------------------
__global__ void s4_bmt(const float* em, float* BmT) {
  int idx = blockIdx.x * blockDim.x + threadIdx.x;
  if (idx >= SY * LNP) return;
  int s = idx / LNP, j = idx % LNP;
  float v;
  if (j >= NST) v = 0.f;
  else if (j == TMI) v = (s == SY - 1) ? 1.f : 0.f;          // terminal one-hot
  else if (j >= 1 && j <= LDIM) {                            // match emission softmax
    if (s == SY - 1) v = 0.f;
    else {
      const float* ek = em + (size_t)(j - 1) * (SY - 1);
      float mx = ek[0];
      for (int q = 1; q < SY - 1; ++q) mx = fmaxf(mx, ek[q]);
      float z = 0.f;
      for (int q = 0; q < SY - 1; ++q) z += expf(ek[q] - mx);
      v = expf(ek[s] - mx) / z;
    }
  } else v = (s < SY - 1) ? (1.0f / (SY - 1)) : 0.f;         // flanks/inserts uniform
  BmT[idx] = v;
}

// ---------------------------------------------------------------------------
// S5: initial distribution
// ---------------------------------------------------------------------------
__global__ void s5_init(Probs P, const float* flank, float* initv) {
  __shared__ float ssum;
  int tid = threadIdx.x;
  if (tid == 0) ssum = 0.f;
  __syncthreads();
  float part = 0.f;
  for (int j = tid; j < LNP; j += 512) {
    float v = (j == 0) ? 0.f : Aval(P, 0, j);
    initv[j] = v;
    part += v;
  }
  atomicAdd(&ssum, part);
  __syncthreads();
  float p0 = 1.0f / (1.0f + expf(-flank[0]));
  float sc = (1.0f - p0) / ssum;
  for (int j = tid; j < LNP; j += 512)
    initv[j] = (j == 0) ? p0 : initv[j] * sc;
}

// ---------------------------------------------------------------------------
// Forward kernel: persistent, 16 WGs x 512 threads, 16 sequences per WG.
// alpha kept in LDS in bf16 WMMA A-operand layout (lane = batch row), dbl-buf.
// Per step per wave: 33 k-tiles x 5 n-tiles wmma_f32_16x16x32_bf16.
// ---------------------------------------------------------------------------
__global__ __launch_bounds__(512, 1)
void hmm_forward(const __bf16* __restrict__ Atile,
                 const float* __restrict__ BmTg,
                 const float* __restrict__ initv,
                 const int* __restrict__ seq,
                 float* __restrict__ out, int T) {
  __shared__ float s_BmT[SY * LNP];                                   // 109,824 B
  __shared__ __attribute__((aligned(32))) __bf16 s_aop[2][NKT * 512]; //  67,584 B
  __shared__ float s_rowsum[16], s_inv[16], s_ll[16];
  __shared__ int   s_sym[16];

  const int tid  = threadIdx.x;
  const int lane = tid & 31;
  const int w    = tid >> 5;           // wave 0..15
  const int b0   = blockIdx.x * 16;

  for (int i = tid; i < SY * LNP; i += 512) s_BmT[i] = BmTg[i];
  if (tid < 16) {
    s_rowsum[tid] = 0.f; s_ll[tid] = 0.f;
    s_sym[tid] = seq[(size_t)(b0 + tid) * T];
  }
  __syncthreads();

  // ----- t = 0 : u0 = init * E0, scatter into A-operand layout ------------
  {
    int m = tid >> 5;                  // 32 threads per row
    int c0 = (tid & 31) * (LNP / 32);
    float lsum = 0.f;
    const float* bt = &s_BmT[s_sym[m] * LNP];
    for (int q = 0; q < LNP / 32; ++q) {
      int j = c0 + q;
      float v = initv[j] * bt[j];
      lsum += v;
      int kt = j >> 5, Kin = j & 31;
      int grp = Kin >> 4, half = (Kin >> 3) & 1, krem = Kin & 7;
      s_aop[0][kt * 512 + (half * 16 + m) * 16 + grp * 8 + krem] = (__bf16)v;
    }
    atomicAdd(&s_rowsum[m], lsum);
  }
  __syncthreads();
  if (tid < 16) {
    float s = s_rowsum[tid];
    s_inv[tid] = 1.0f / s; s_ll[tid] += logf(s); s_rowsum[tid] = 0.f;
  }
  __syncthreads();
  for (int i = tid; i < NKT * 512; i += 512) {
    int m = (i >> 4) & 15;             // lane%16 encodes batch row
    s_aop[0][i] = (__bf16)((float)s_aop[0][i] * s_inv[m]);
  }
  __syncthreads();

  // ----- scan over t = 1..T-1 ---------------------------------------------
  int cur = 0;
  for (int t = 1; t < T; ++t) {
    int nxt = cur ^ 1;
    if (tid < 16) s_sym[tid] = seq[(size_t)(b0 + tid) * T + t];
    __syncthreads();

    // GEMM: acc[q] (16x16 f32) over 33 k-tiles, 5 n-tiles per wave
    v8f acc[5] = {};
    for (int kt = 0; kt < NKT; ++kt) {
      v16bf a = *(const v16bf*)&s_aop[cur][kt * 512 + lane * 16];
      const __bf16* tb = Atile + (size_t)kt * NNT * 512 + (size_t)w * 512 + (size_t)lane * 16;
#pragma unroll
      for (int q = 0; q < 5; ++q) {
        v16bf b = *(const v16bf*)(tb + (size_t)q * 16 * 512);
        acc[q] = __builtin_amdgcn_wmma_f32_16x16x32_bf16(
            false, a, false, b, (short)0, acc[q], false, false);
      }
    }

    // Emission multiply + row sums + scatter-store next alpha (bf16, A-layout)
    int n_in = lane & 15;
    int mh = (lane >> 4) * 8;          // C layout: lanes16-31 hold rows M=8..15
#pragma unroll
    for (int q = 0; q < 5; ++q) {
      int nt = w + 16 * q;
      if (nt < NNT_REAL) {             // wave-uniform guard for padded n-tiles
        int j = nt * 16 + n_in;
        int kt2 = j >> 5, Kin = j & 31;
        int grp = Kin >> 4, half = (Kin >> 3) & 1, krem = Kin & 7;
#pragma unroll
        for (int r = 0; r < 8; ++r) {
          int m = mh + r;
          float v = acc[q][r] * s_BmT[s_sym[m] * LNP + j];
          s_aop[nxt][kt2 * 512 + (half * 16 + m) * 16 + grp * 8 + krem] = (__bf16)v;
          float rs = v;                // reduce over 16 cols (xor stays in half)
          rs += __shfl_xor(rs, 1, 32);
          rs += __shfl_xor(rs, 2, 32);
          rs += __shfl_xor(rs, 4, 32);
          rs += __shfl_xor(rs, 8, 32);
          if (n_in == 0) atomicAdd(&s_rowsum[m], rs);   // ds_add_f32
        }
      }
    }
    __syncthreads();
    if (tid < 16) {
      float s = s_rowsum[tid];
      s_inv[tid] = 1.0f / s; s_ll[tid] += logf(s); s_rowsum[tid] = 0.f;
    }
    __syncthreads();
    for (int i = tid; i < NKT * 512; i += 512) {
      int m = (i >> 4) & 15;
      s_aop[nxt][i] = (__bf16)((float)s_aop[nxt][i] * s_inv[m]);
    }
    __syncthreads();
    cur = nxt;
  }

  if (tid < 16) out[b0 + tid] = s_ll[tid];
}

// ---------------------------------------------------------------------------
extern "C" void kernel_launch(void* const* d_in, const int* in_sizes, int n_in,
                              void* d_out, int out_size, void* d_ws, size_t ws_size,
                              hipStream_t stream) {
  const float* em   = (const float*)d_in[0];
  const float* b2m  = (const float*)d_in[1];
  const float* m2e  = (const float*)d_in[2];
  const float* m2m  = (const float*)d_in[3];
  const float* m2i  = (const float*)d_in[4];
  const float* i2m  = (const float*)d_in[5];
  const float* i2i  = (const float*)d_in[6];
  const float* md   = (const float*)d_in[7];
  const float* d2m  = (const float*)d_in[8];
  const float* d2d  = (const float*)d_in[9];
  const float* lfl  = (const float*)d_in[10];
  const float* lfe  = (const float*)d_in[11];
  const float* e2u  = (const float*)d_in[12];
  const float* e2r  = (const float*)d_in[13];
  const float* e2t  = (const float*)d_in[14];
  const float* flk  = (const float*)d_in[15];
  const int*   seq  = (const int*)d_in[16];
  float* out = (float*)d_out;

  const int B = out_size;                  // 256
  const int T = in_sizes[16] / B;          // 400

  char* ws = (char*)d_ws;
  size_t off = 0;
  auto alloc = [&](size_t bytes) -> void* {
    void* p = ws + off;
    off = (off + bytes + 255) & ~(size_t)255;
    return p;
  };
  float* P_b2m  = (float*)alloc(LDIM * 4);
  float* P_m2e  = (float*)alloc(LDIM * 4);
  float* P_m2m  = (float*)alloc(LDIM * 4);
  float* P_m2i  = (float*)alloc(LDIM * 4);
  float* P_i2m  = (float*)alloc(LDIM * 4);
  float* P_i2i  = (float*)alloc(LDIM * 4);
  float* P_md   = (float*)alloc(LDIM * 4);
  float* P_d2m  = (float*)alloc(LDIM * 4);
  float* P_d2d  = (float*)alloc(LDIM * 4);
  float* cp     = (float*)alloc(LDIM * 4);
  float* entry  = (float*)alloc(LDIM * 4);
  float* exitp  = (float*)alloc(LDIM * 4);
  float* scal   = (float*)alloc(16 * 4);
  float* initv  = (float*)alloc(LNP * 4);
  float* BmT    = (float*)alloc((size_t)SY * LNP * 4);
  __bf16* Atile = (__bf16*)alloc((size_t)NKT * NNT * 512 * 2);

  Probs P = { P_b2m, P_m2e, P_m2m, P_m2i, P_i2m, P_i2i, P_md, P_d2m,
              cp, entry, exitp, scal };

  s1_probs<<<1, 512, 0, stream>>>(b2m, m2e, m2m, m2i, i2m, i2i, md, d2m, d2d,
                                  lfl, lfe, e2u, e2r, e2t,
                                  P_b2m, P_m2e, P_m2m, P_m2i, P_i2m, P_i2i,
                                  P_md, P_d2m, P_d2d, scal);
  s2_scan<<<1, 1, 0, stream>>>(P_md, P_d2m, P_d2d, P_b2m, P_m2e,
                               cp, entry, exitp, scal);
  s3_atile<<<dim3(NNT, NKT), 512, 0, stream>>>(P, Atile);
  s4_bmt<<<(SY * LNP + 511) / 512, 512, 0, stream>>>(em, BmT);
  s5_init<<<1, 512, 0, stream>>>(P, flk, initv);
  hmm_forward<<<B / 16, 512, 0, stream>>>(Atile, BmT, initv, seq, out, T);
}
